// MultiHeadAttention_4458176053709
// MI455X (gfx1250) — matrix operand order repaired and checked by reading, not compiled, not run on hardware
//
#include <hip/hip_runtime.h>
#include <hip/hip_bf16.h>
#include <stdint.h>

// ---------------------------------------------------------------------------
// CDNA5 (gfx1250) wave32 WMMA f16 MHA forward, TDM-staged attention tiles.
// ---------------------------------------------------------------------------

typedef __attribute__((ext_vector_type(16))) _Float16 v16h;
typedef __attribute__((ext_vector_type(8)))  float    v8f;

union FragH { v16h v; _Float16 h[16]; uint32_t u[8]; };
union FragF { v8f  v; float    f[8]; };

// A-matrix 16x32 f16 fragment (ISA 7.12.2): lane L -> M = L&15;
// VGPR j (pair lo/hi): K = (j<4 ? 2j : 16+2(j-4)) + (L<16 ? 0 : 8)
__device__ inline FragH load_frag_a16(const _Float16* base, int stride, int lane) {
  FragH f;
  const int m  = lane & 15;
  const int kb = (lane & 16) ? 8 : 0;
  const _Float16* row = base + m * stride;
#pragma unroll
  for (int j = 0; j < 8; ++j) {
    const int k = ((j & 4) ? 16 : 0) + kb + 2 * (j & 3);
    f.u[j] = *(const uint32_t*)(row + k);          // pair load (k, k+1)
  }
  return f;
}

// B-matrix 32x16 f16 fragment: lane L -> N = L&15;
// VGPR j (pair lo/hi): K = (j<4 ? 2j : 16+2(j-4)) + (L<16 ? 0 : 8).
// `base` points at an LDS tile stored K-contiguous per N row: Bs[n][k].
__device__ inline FragH load_frag_b16(const _Float16* base, int stride, int lane) {
  FragH f;
  const int n  = lane & 15;
  const int kb = (lane & 16) ? 8 : 0;
  const _Float16* row = base + n * stride + kb;
#pragma unroll
  for (int j = 0; j < 8; ++j) {
    f.u[j] = *(const uint32_t*)(row + ((j & 4) ? 16 : 0) + 2 * (j & 3));      // pair load (k, k+1)
  }
  return f;
}

// ---------------------------------------------------------------------------
// Tensor Data Mover: 2D f16 tile, global -> LDS, with LDS padding of 1 DWORD
// after every 32 DWORDs (i.e. dest row stride = 64 halfs + 2 = 66 halfs).
// D# layout per CDNA5 ISA 8.3/8.4. clang-23 6-arg builtin form.
// ---------------------------------------------------------------------------
#if defined(__HIP_DEVICE_COMPILE__) && __has_builtin(__builtin_amdgcn_tensor_load_to_lds)
#define HAVE_TDM 1
typedef __attribute__((ext_vector_type(4))) uint32_t tdm_u32x4;
typedef __attribute__((ext_vector_type(8))) int      tdm_i32x8;
typedef __attribute__((ext_vector_type(4))) int      tdm_i32x4;

// Loads a (rows=64) x (cols=64) f16 tile whose rows are `row_stride_elems`
// apart in global memory into LDS at byte offset lds_off, dest stride 66 halfs.
__device__ inline void tdm_load_tile_64x64_f16(uint32_t lds_off, const void* gptr,
                                               uint32_t row_stride_elems) {
  const uint64_t ga = (uint64_t)(uintptr_t)gptr;
  tdm_u32x4 g0;
  g0[0] = 1u;                                   // count=1, user mode, no gather
  g0[1] = lds_off;                              // lds_addr (bytes)
  g0[2] = (uint32_t)ga;                         // global_addr[31:0]
  g0[3] = (uint32_t)(ga >> 32) | (2u << 30);    // global_addr[56:32] | type=2
  tdm_i32x8 g1;
  // w0: wg_mask=0 | data_size=1(2B)<<16 | pad_enable<<20 | pad_interval=4(32DW)<<22
  //     | pad_amount=0(1DW)<<25
  g1[0] = (int)((1u << 16) | (1u << 20) | (4u << 22));
  g1[1] = (int)(64u << 16);                     // tensor_dim0[15:0]=64 at bits 63:48
  g1[2] = (int)(64u << 16);                     // tensor_dim1[15:0]=64 at bits 95:80
  g1[3] = (int)(64u << 16);                     // tile_dim0=64 at bits 127:112
  g1[4] = (int)64;                              // tile_dim1=64; tile_dim2=0
  g1[5] = (int)row_stride_elems;                // tensor_dim0_stride[31:0]
  g1[6] = 0;                                    // stride hi, tensor_dim1_stride lo
  g1[7] = 0;
  const tdm_i32x4 z4 = {0, 0, 0, 0};
  const tdm_i32x8 z8 = {0, 0, 0, 0, 0, 0, 0, 0};
  __builtin_amdgcn_tensor_load_to_lds(g0, g1, z4, z4, z8, 0);
}
#else
#define HAVE_TDM 0
#endif

// ---------------------------------------------------------------------------
// GEMM: Out[M,N] = (A[M,K] @ W[K,N] + bias) * out_scale
// Block tile 128x128, K-step 64. 256 threads = 8 waves (4 in M x 2 in N),
// each wave owns a 32x64 output patch = 2x4 WMMA C fragments.
// ---------------------------------------------------------------------------
#define GT_M 128
#define GT_N 128
#define GT_K 64
#define GA_STR (GT_K + 2)   // 66 halfs: even (4B pair loads ok), conflict-breaking
#define GB_STR (GT_K + 2)

template <bool A_HALF, bool OUT_HALF>
__global__ __launch_bounds__(256)
void mha_gemm_kernel(const void* __restrict__ Ain, const float* __restrict__ W,
                     const float* __restrict__ bias, void* __restrict__ Out,
                     int M, int N, int K, float out_scale) {
  __shared__ _Float16 As[GT_M * GA_STR];
  __shared__ _Float16 Bs[GT_N * GB_STR];   // stored transposed: Bs[n][k]

  const int tid   = threadIdx.x;
  const int lane  = tid & 31;
  const int wave  = tid >> 5;
  const int waveM = wave & 3;
  const int waveN = wave >> 2;
  const long block_m = (long)blockIdx.x * GT_M;
  const long block_n = (long)blockIdx.y * GT_N;

  FragF acc[2][4];
#pragma unroll
  for (int mt = 0; mt < 2; ++mt)
#pragma unroll
    for (int nt = 0; nt < 4; ++nt)
#pragma unroll
      for (int j = 0; j < 8; ++j) acc[mt][nt].f[j] = 0.0f;

  for (int k0 = 0; k0 < K; k0 += GT_K) {
    // ---- stage A tile (128 x 64) into LDS as f16 ----
    if constexpr (A_HALF) {
      const _Float16* A = (const _Float16*)Ain;
#pragma unroll
      for (int i = 0; i < 16; ++i) {               // 4096 u32 / 256 threads
        const int idx = tid + i * 256;
        const int r = idx >> 5;
        const int c = (idx & 31) * 2;
        *(uint32_t*)(As + r * GA_STR + c) =
            *(const uint32_t*)(A + (block_m + r) * (long)K + k0 + c);
      }
      if (k0 + GT_K < K)                           // prefetch next A tile
        __builtin_prefetch(A + (block_m + (tid >> 1)) * (long)K + k0 + GT_K +
                               (tid & 1) * 32, 0, 1);
    } else {
      const float* A = (const float*)Ain;
#pragma unroll
      for (int i = 0; i < 8; ++i) {                // 2048 float4 / 256 threads
        const int idx = tid + i * 256;
        const int r = idx >> 4;
        const int c = (idx & 15) * 4;
        const float4 v = *(const float4*)(A + (block_m + r) * (long)K + k0 + c);
        _Float16* d = As + r * GA_STR + c;
        d[0] = (_Float16)v.x; d[1] = (_Float16)v.y;
        d[2] = (_Float16)v.z; d[3] = (_Float16)v.w;
      }
      if (k0 + GT_K < K)                           // prefetch next A tile
        __builtin_prefetch(A + (block_m + (tid >> 1)) * (long)K + k0 + GT_K +
                               (tid & 1) * 32, 0, 1);
    }
    // ---- stage W tile (64 x 128) transposed into Bs[n][k] ----
#pragma unroll
    for (int i = 0; i < 8; ++i) {
      const int idx = tid + i * 256;
      const int kr = idx >> 5;
      const int c  = (idx & 31) * 4;
      const float4 v = *(const float4*)(W + (long)(k0 + kr) * N + block_n + c);
      Bs[(c + 0) * GB_STR + kr] = (_Float16)v.x;
      Bs[(c + 1) * GB_STR + kr] = (_Float16)v.y;
      Bs[(c + 2) * GB_STR + kr] = (_Float16)v.z;
      Bs[(c + 3) * GB_STR + kr] = (_Float16)v.w;
    }
    if (k0 + GT_K < K)                             // prefetch next W tile
      __builtin_prefetch(W + (long)(k0 + GT_K + (tid >> 2)) * N + block_n +
                             (tid & 3) * 32, 0, 1);
    __syncthreads();

#pragma unroll
    for (int kk = 0; kk < GT_K; kk += 32) {
      FragH a[2], b[4];
#pragma unroll
      for (int mt = 0; mt < 2; ++mt)
        a[mt] = load_frag_a16(As + (waveM * 32 + mt * 16) * GA_STR + kk, GA_STR, lane);
#pragma unroll
      for (int nt = 0; nt < 4; ++nt)
        b[nt] = load_frag_b16(Bs + (waveN * 64 + nt * 16) * GB_STR + kk, GB_STR, lane);
#pragma unroll
      for (int mt = 0; mt < 2; ++mt)
#pragma unroll
        for (int nt = 0; nt < 4; ++nt)
          acc[mt][nt].v = __builtin_amdgcn_wmma_f32_16x16x32_f16(
              false, a[mt].v, false, b[nt].v, (short)0, acc[mt][nt].v, false, false);
    }
    __syncthreads();
  }

  // ---- epilogue: C layout lane=N, VGPR j -> M = j (+8 for upper lane half) ----
  const int n_in = lane & 15;
  const int mh   = (lane & 16) ? 8 : 0;
#pragma unroll
  for (int mt = 0; mt < 2; ++mt)
#pragma unroll
    for (int nt = 0; nt < 4; ++nt) {
      const long col = block_n + waveN * 64 + nt * 16 + n_in;
      const float bv = bias[col];
#pragma unroll
      for (int j = 0; j < 8; ++j) {
        const long row = block_m + waveM * 32 + mt * 16 + mh + j;
        const float val = (acc[mt][nt].f[j] + bv) * out_scale;
        if constexpr (OUT_HALF)
          ((_Float16*)Out)[row * N + col] = (_Float16)val;
        else
          ((float*)Out)[row * N + col] = val;
      }
    }
}

// ---------------------------------------------------------------------------
// Flash attention: 1 block = 128 query rows of one (b,h). 8 waves x 16 queries.
// dh = 64. K chunk staged by the Tensor Data Mover (TDM) when available;
// V chunk staged transposed by VALU; online softmax; P@V in f32 C fragments.
// 1/sqrt(Dk) already folded into the Q projection.
// ---------------------------------------------------------------------------
#define KV_STR 66
#define PS_STR 66

__global__ __launch_bounds__(256)
void mha_attn_kernel(const _Float16* __restrict__ Qp, const _Float16* __restrict__ Kp,
                     const _Float16* __restrict__ Vp, _Float16* __restrict__ Ow, int S) {
  __shared__ _Float16 Ks[64 * KV_STR];          // Ks[key][dh]  (B-frag: n=key, k=dh)
  __shared__ _Float16 Vs[64 * KV_STR];          // Vs[dh][key]  (B-frag: n=dh,  k=key)
  __shared__ _Float16 Ps[8 * 16 * PS_STR];      // per-wave P staging: Ps[q][key]

  const int tid  = threadIdx.x;
  const int lane = tid & 31;
  const int wave = tid >> 5;
  const int h = blockIdx.y, b = blockIdx.z;
  const long rowstride = 1024;
  const long base = (long)b * S * rowstride + (long)h * 64;
  const int q0 = blockIdx.x * 128 + wave * 16;

  // Q A-fragments are loop-invariant: load once straight from global.
  FragH aq[2];
  {
    const int m  = lane & 15;
    const int kb = (lane & 16) ? 8 : 0;
    const _Float16* qrow = Qp + base + (long)(q0 + m) * rowstride;
#pragma unroll
    for (int hf = 0; hf < 2; ++hf)
#pragma unroll
      for (int j = 0; j < 8; ++j) {
        const int k = hf * 32 + ((j & 4) ? 16 : 0) + kb + 2 * (j & 3);
        aq[hf].u[j] = *(const uint32_t*)(qrow + k);
      }
  }

  float m_i[8], l_i[8];
  FragF oacc[4];
#pragma unroll
  for (int j = 0; j < 8; ++j) { m_i[j] = -1e30f; l_i[j] = 0.0f; }
#pragma unroll
  for (int dt = 0; dt < 4; ++dt)
#pragma unroll
    for (int j = 0; j < 8; ++j) oacc[dt].f[j] = 0.0f;

  for (int kt = 0; kt < S; kt += 64) {
    // ---- stage 64 keys into Ks[key][dh] ----
#if HAVE_TDM
    if (wave == 0) {
      tdm_load_tile_64x64_f16((uint32_t)(uintptr_t)Ks,
                              Kp + base + (long)kt * rowstride, 1024u);
    }
#else
#pragma unroll
    for (int i = 0; i < 8; ++i) {
      const int idx = tid + i * 256;
      const int r = idx >> 5;
      const int c = (idx & 31) * 2;
      *(uint32_t*)(Ks + r * KV_STR + c) =
          *(const uint32_t*)(Kp + base + (long)(kt + r) * rowstride + c);
    }
#endif
    // ---- stage 64 values transposed into Vs[dh][key] ----
#pragma unroll
    for (int i = 0; i < 8; ++i) {
      const int idx = tid + i * 256;
      const int r = idx >> 5;
      const int c = (idx & 31) * 2;
      const _Float16* vsrc = Vp + base + (long)(kt + r) * rowstride + c;
      Vs[(c + 0) * KV_STR + r] = vsrc[0];
      Vs[(c + 1) * KV_STR + r] = vsrc[1];
    }
    if (kt + 64 < S) {                             // prefetch next K/V chunk
      __builtin_prefetch(Kp + base + (long)(kt + 64 + (tid >> 2)) * rowstride, 0, 1);
      __builtin_prefetch(Vp + base + (long)(kt + 64 + (tid >> 2)) * rowstride, 0, 1);
    }
#if HAVE_TDM
    if (wave == 0) __builtin_amdgcn_s_wait_tensorcnt(0);
#endif
    __syncthreads();

    // ---- S = Q K^T (16 queries x 64 keys): 2 K-steps x 4 key tiles ----
    FragF s[4];
#pragma unroll
    for (int nt = 0; nt < 4; ++nt)
#pragma unroll
      for (int j = 0; j < 8; ++j) s[nt].f[j] = 0.0f;
#pragma unroll
    for (int kk = 0; kk < 2; ++kk)
#pragma unroll
      for (int nt = 0; nt < 4; ++nt) {
        FragH bk = load_frag_b16(Ks + (nt * 16) * KV_STR + kk * 32, KV_STR, lane);
        s[nt].v = __builtin_amdgcn_wmma_f32_16x16x32_f16(
            false, aq[kk].v, false, bk.v, (short)0, s[nt].v, false, false);
      }

    // ---- online softmax: row M=(half,j) spans the 16 lanes of this half ----
#pragma unroll
    for (int j = 0; j < 8; ++j) {
      float rm = fmaxf(fmaxf(s[0].f[j], s[1].f[j]), fmaxf(s[2].f[j], s[3].f[j]));
#pragma unroll
      for (int off = 1; off < 16; off <<= 1)
        rm = fmaxf(rm, __shfl_xor(rm, off, 32));
      const float mnew = fmaxf(m_i[j], rm);
      const float corr = __expf(m_i[j] - mnew);
      m_i[j] = mnew;
      float rs = 0.0f;
#pragma unroll
      for (int nt = 0; nt < 4; ++nt) {
        const float p = __expf(s[nt].f[j] - mnew);
        s[nt].f[j] = p;
        rs += p;
      }
#pragma unroll
      for (int off = 1; off < 16; off <<= 1)
        rs += __shfl_xor(rs, off, 32);
      l_i[j] = l_i[j] * corr + rs;
#pragma unroll
      for (int dt = 0; dt < 4; ++dt) oacc[dt].f[j] *= corr;
    }

    // ---- reshape P: C-layout -> A-layout via per-wave LDS slab (same-wave,
    //      DS ops in-order; no cross-wave barrier needed) ----
    _Float16* ps = Ps + wave * (16 * PS_STR);
    {
      const int n  = lane & 15;
      const int mh = (lane & 16) ? 8 : 0;
#pragma unroll
      for (int j = 0; j < 8; ++j)
#pragma unroll
        for (int nt = 0; nt < 4; ++nt)
          ps[(mh + j) * PS_STR + nt * 16 + n] = (_Float16)s[nt].f[j];
    }

    // ---- O += P @ V : 2 K-steps (64 keys) x 4 dh tiles ----
#pragma unroll
    for (int kk = 0; kk < 2; ++kk) {
      FragH ap = load_frag_a16(ps + kk * 32, PS_STR, lane);
#pragma unroll
      for (int dt = 0; dt < 4; ++dt) {
        FragH bv = load_frag_b16(Vs + (dt * 16) * KV_STR + kk * 32, KV_STR, lane);
        oacc[dt].v = __builtin_amdgcn_wmma_f32_16x16x32_f16(
            false, ap.v, false, bv.v, (short)0, oacc[dt].v, false, false);
      }
    }
    __syncthreads();
  }

  // ---- epilogue: O = acc / l_i -> f16, merged-heads layout (b, s, h*64+dh) ----
  const int n  = lane & 15;
  const int mh = (lane & 16) ? 8 : 0;
#pragma unroll
  for (int dt = 0; dt < 4; ++dt)
#pragma unroll
    for (int j = 0; j < 8; ++j) {
      const long row = q0 + mh + j;
      const float val = oacc[dt].f[j] / l_i[j];
      Ow[base + row * rowstride + dt * 16 + n] = (_Float16)val;
    }
}

// ---------------------------------------------------------------------------
extern "C" void kernel_launch(void* const* d_in, const int* in_sizes, int n_in,
                              void* d_out, int out_size, void* d_ws, size_t ws_size,
                              hipStream_t stream) {
  (void)in_sizes; (void)n_in; (void)out_size; (void)ws_size;
  const float* queries = (const float*)d_in[0];
  const float* keys    = (const float*)d_in[1];
  const float* values  = (const float*)d_in[2];
  const float* Wq = (const float*)d_in[3];
  const float* bq = (const float*)d_in[4];
  const float* Wk = (const float*)d_in[5];
  const float* bk = (const float*)d_in[6];
  const float* Wv = (const float*)d_in[7];
  const float* bv = (const float*)d_in[8];
  const float* Wo = (const float*)d_in[9];
  const float* bo = (const float*)d_in[10];
  float* out = (float*)d_out;

  const int B = 4, S = 2048, D = 1024;
  const int M = B * S;                       // 8192
  _Float16* Qp = (_Float16*)d_ws;            // each 8192*1024 halfs = 16 MB
  _Float16* Kp = Qp + (size_t)M * D;
  _Float16* Vp = Kp + (size_t)M * D;
  _Float16* Ow = Vp + (size_t)M * D;         // total 64 MB workspace

  dim3 ggrid(M / GT_M, D / GT_N);            // 64 x 8

  // Fold 1/sqrt(D_K)=1/32 into the Q projection output.
  mha_gemm_kernel<false, true><<<ggrid, 256, 0, stream>>>(queries, Wq, bq, Qp, M, D, D, 0.03125f);
  mha_gemm_kernel<false, true><<<ggrid, 256, 0, stream>>>(keys,    Wk, bk, Kp, M, D, D, 1.0f);
  mha_gemm_kernel<false, true><<<ggrid, 256, 0, stream>>>(values,  Wv, bv, Vp, M, D, D, 1.0f);

  mha_attn_kernel<<<dim3(S / 128, 16, B), 256, 0, stream>>>(Qp, Kp, Vp, Ow, S);

  mha_gemm_kernel<true, false><<<ggrid, 256, 0, stream>>>(Ow, Wo, bo, out, M, D, D, 1.0f);
}
